// PageNet_40175124086806
// MI455X (gfx1250) — matrix operand order repaired and checked by reading, not compiled, not run on hardware
//
#include <hip/hip_runtime.h>

typedef unsigned short u16;
typedef unsigned int   u32;
typedef __attribute__((ext_vector_type(16))) __bf16 v16bf;
typedef __attribute__((ext_vector_type(8)))  float  v8f;

struct __align__(16) Q4 { u32 x, y, z, w; };
union FragAB { v16bf v; Q4 q[2]; u16 u[16]; };

__device__ __forceinline__ Q4 zero_q4() { Q4 z; z.x = z.y = z.z = z.w = 0u; return z; }

__device__ __forceinline__ u16 f2bf(float f) {
  u32 u = __float_as_uint(f);
  u32 r = u + 0x7fffu + ((u >> 16) & 1u);   // round-to-nearest-even
  return (u16)(r >> 16);
}
__device__ __forceinline__ float bf2f(u16 h) { return __uint_as_float(((u32)h) << 16); }

__device__ __forceinline__ v8f wmma_bf16(v16bf a, v16bf b, v8f c) {
  // D = A(16x32 bf16) * B(32x16 bf16) + C(16x16 f32)
  return __builtin_amdgcn_wmma_f32_16x16x32_bf16(false, a, false, b, (short)0, c, false, false);
}
__device__ __forceinline__ float hswish(float v) {
  return v * fminf(fmaxf(v + 3.0f, 0.0f), 6.0f) * (1.0f / 6.0f);
}

// ---------------------------------------------------------------------------
// feature: NCHW f32 [8,512,128,128] -> NHWC bf16 via LDS tile transpose
// ---------------------------------------------------------------------------
__global__ __launch_bounds__(256) void k_cvt_feature(const float* __restrict__ F32,
                                                     u16* __restrict__ Fb) {
  __shared__ float tile[32][33];
  int blk = blockIdx.x;                 // 8 * 512 * 16 blocks
  int ct = blk & 15;
  int pt = (blk >> 4) & 511;
  int nb = blk >> 13;
  int c0 = ct * 32, p0 = pt * 32;
  int lx = threadIdx.x & 31;
  int ly = threadIdx.x >> 5;            // 0..7
  for (int r = 0; r < 32; r += 8) {
    int c = c0 + ly + r;
    tile[ly + r][lx] = F32[(((size_t)nb * 512 + c) << 14) + p0 + lx];
  }
  __syncthreads();
  for (int r = 0; r < 32; r += 8) {
    int p = p0 + ly + r;
    Fb[((size_t)(nb * 16384 + p)) * 512 + c0 + lx] = f2bf(tile[lx][ly + r]);
  }
}

// ---------------------------------------------------------------------------
// Pack 3x3 conv weights [Cout,Cin,3,3] f32 -> [Cout][K=(ky*3+kx)*Cin+cin] bf16
// ---------------------------------------------------------------------------
__global__ __launch_bounds__(256) void k_pack_w3x3(const float* __restrict__ w,
                                                   u16* __restrict__ o, int Cout, int Cin) {
  int Kc = Cin * 9;
  int idx = blockIdx.x * blockDim.x + threadIdx.x;
  if (idx >= Cout * Kc) return;
  int co = idx / Kc, k = idx - co * Kc;
  int tap = k / Cin, cin = k - tap * Cin;
  o[idx] = f2bf(w[(size_t)(co * Cin + cin) * 9 + tap]);
}

__global__ __launch_bounds__(256) void k_cvt_bf(const float* __restrict__ x,
                                                u16* __restrict__ y, int n) {
  int i = blockIdx.x * blockDim.x + threadIdx.x;
  if (i < n) y[i] = f2bf(x[i]);
}

// ---------------------------------------------------------------------------
// ConvTranspose tap helper (parity decomposition of k3/s2/p1/op1)
// ---------------------------------------------------------------------------
__device__ __forceinline__ void convt_tap(int p, int tap, int& ky, int& kx) {
  if (p == 0)      { ky = 1;                  kx = 1; }
  else if (p == 1) { ky = 1;                  kx = tap ? 2 : 0; }
  else if (p == 2) { ky = tap ? 2 : 0;        kx = 1; }
  else             { ky = (tap >> 1) ? 2 : 0; kx = (tap & 1) ? 2 : 0; }
}

// Pack convT weights: per parity p, B[row<32:rg | row<40:hm | pad][K=tap*512+cin]
// wk[ky][kx][cout][cin] = w_orig[cin][cout][2-ky][2-kx]
__global__ __launch_bounds__(256) void k_pack_bt(const float* __restrict__ rgw,
                                                 const float* __restrict__ hmw,
                                                 u16* __restrict__ BT) {
  int idx = blockIdx.x * blockDim.x + threadIdx.x;
  if (idx >= 48 * 4608) return;
  int pval, local;
  if (idx < 24576)       { pval = 0; local = idx; }
  else if (idx < 73728)  { pval = 1; local = idx - 24576; }
  else if (idx < 122880) { pval = 2; local = idx - 73728; }
  else                   { pval = 3; local = idx - 122880; }
  int Kp = (1 << ((pval + 1) >> 1)) << 9;      // 512,1024,1024,2048
  int row = local / Kp;
  int k = local - row * Kp;
  int tap = k >> 9, cin = k & 511;
  int ky, kx; convt_tap(pval, tap, ky, kx);
  int sy = 2 - ky, sx = 2 - kx;
  float v = 0.0f;
  if (row < 32)      v = rgw[((size_t)(cin * 32 + row) * 3 + sy) * 3 + sx];
  else if (row < 40) v = hmw[((size_t)(cin * 8 + (row - 32)) * 3 + sy) * 3 + sx];
  BT[idx] = f2bf(v);
}

// ---------------------------------------------------------------------------
// Generic implicit-GEMM conv (k in {1,3}, stride in {1,2}), bf16 WMMA.
// X: NHWC bf16, W: [Cout][K] bf16, Y: raw conv+bias, NHWC f32.
// Wave computes 32(M) x 64(N): two A fragments share every B fragment.
// Tap-outer K loop keeps the inner loop free of divisions.
// Requires Cin % 32 == 0, Cout % 16 == 0, (N*Hout*Wout) % 32 == 0.
// ---------------------------------------------------------------------------
__global__ __launch_bounds__(256) void k_conv_wmma(
    const u16* __restrict__ X, const u16* __restrict__ W,
    const float* __restrict__ bias, float* __restrict__ Y,
    int N, int Hin, int Win, int Cin, int Hout, int Wout, int Cout,
    int ksize, int stride, int pad) {
  const int Kc = Cin * ksize * ksize;
  const int HWo = Hout * Wout;
  const int ngroups = (Cout + 63) >> 6;
  const int mtiles = (N * HWo) >> 5;
  const int lane = threadIdx.x & 31;
  const int wid = blockIdx.x * (blockDim.x >> 5) + (threadIdx.x >> 5);
  if (wid >= mtiles * ngroups) return;
  const int ng = wid % ngroups;
  const int mtile = wid / ngroups;

  const int arow = lane & 15;
  const int half = lane >> 4;
  const int kg = half * 8;
  const int bcol = lane & 15;
  const int bko = half * 8;

  int nn[2], iy0[2], ix0[2];
#pragma unroll
  for (int h2 = 0; h2 < 2; ++h2) {
    const int m = mtile * 32 + h2 * 16 + arow;
    const int n = m / HWo;
    const int rem = m - n * HWo;
    const int oy = rem / Wout;
    const int ox = rem - oy * Wout;
    nn[h2] = n;
    iy0[h2] = oy * stride - pad;
    ix0[h2] = ox * stride - pad;
  }

  v8f acc[2][4];
#pragma unroll
  for (int h2 = 0; h2 < 2; ++h2)
#pragma unroll
    for (int t = 0; t < 4; ++t) acc[h2][t] = v8f{0, 0, 0, 0, 0, 0, 0, 0};

  const int ntap = ksize * ksize;
  for (int tap = 0; tap < ntap; ++tap) {
    const int ky = tap / ksize;
    const int kx = tap - ky * ksize;
    const u16* pa[2];
    bool ok[2];
#pragma unroll
    for (int h2 = 0; h2 < 2; ++h2) {
      const int iy = iy0[h2] + ky, ix = ix0[h2] + kx;
      ok[h2] = (iy >= 0) && (iy < Hin) && (ix >= 0) && (ix < Win);
      pa[h2] = X + ((size_t)((nn[h2] * Hin + iy) * Win + ix)) * Cin + kg;
    }
    const int kbase = tap * Cin;
    for (int cin0 = 0; cin0 < Cin; cin0 += 32) {
      FragAB a[2];
#pragma unroll
      for (int h2 = 0; h2 < 2; ++h2) {
        if (ok[h2]) {
          const u16* p = pa[h2] + cin0;
          a[h2].q[0] = *(const Q4*)p;
          a[h2].q[1] = *(const Q4*)(p + 16);
        } else { a[h2].q[0] = zero_q4(); a[h2].q[1] = zero_q4(); }
      }
#pragma unroll
      for (int nt = 0; nt < 4; ++nt) {
        const int cobase = ng * 64 + nt * 16;
        if (cobase >= Cout) break;             // uniform across wave (Cout%16==0)
        FragAB b;
        const u16* pb = W + (size_t)(cobase + bcol) * Kc + kbase + cin0 + bko;
        b.q[0] = *(const Q4*)pb;
        b.q[1] = *(const Q4*)(pb + 16);
        acc[0][nt] = wmma_bf16(a[0].v, b.v, acc[0][nt]);
        acc[1][nt] = wmma_bf16(a[1].v, b.v, acc[1][nt]);
      }
    }
  }
#pragma unroll
  for (int h2 = 0; h2 < 2; ++h2)
#pragma unroll
    for (int nt = 0; nt < 4; ++nt) {
      const int cobase = ng * 64 + nt * 16;
      if (cobase >= Cout) break;
      const int co = cobase + bcol;
      const float bb = bias[co];
#pragma unroll
      for (int v = 0; v < 8; ++v) {
        const int mm = mtile * 32 + h2 * 16 + v + half * 8;
        Y[(size_t)mm * Cout + co] = acc[h2][nt][v] + bb;
      }
    }
}

// ---------------------------------------------------------------------------
// Fused ConvTranspose2d (rg 32ch + hm 8ch share A fragments), parity decomposed.
// X: feature NHWC bf16 [8,128,128,512]; BT packed per-parity [48][Kp] bf16.
// Wave computes 32(M) x 48(N). R: [8,32,256,256] f32 (+rg bias),
// HMo: [8,8,256,256] f32 (+hm bias).
// ---------------------------------------------------------------------------
__global__ __launch_bounds__(256) void k_convt_wmma(
    const u16* __restrict__ X, const u16* __restrict__ BT,
    const float* __restrict__ rgb, const float* __restrict__ hmb,
    float* __restrict__ R, float* __restrict__ HMo) {
  const int lane = threadIdx.x & 31;
  const int wid = blockIdx.x * (blockDim.x >> 5) + (threadIdx.x >> 5);
  const int p = wid >> 12;                      // parity 0..3 (uniform per wave)
  const int mtile = wid & 4095;
  const int ntap = 1 << ((p + 1) >> 1);         // 1,2,2,4
  const int Kp = ntap << 9;                     // 512,1024,1024,2048
  const int offE = (p > 0 ? 48 * 512 : 0) + (p > 1 ? (p - 1) * 48 * 1024 : 0);
  const int py = p >> 1, px = p & 1;

  const int arow = lane & 15;
  const int half = lane >> 4;
  const int kg = half * 8;
  const int bcol = lane & 15;
  const int bko = half * 8;

  int nn[2], ii[2], jj[2];
#pragma unroll
  for (int h2 = 0; h2 < 2; ++h2) {
    const int m = mtile * 32 + h2 * 16 + arow;
    nn[h2] = m >> 14;
    const int rem = m & 16383;
    ii[h2] = rem >> 7;
    jj[h2] = rem & 127;
  }

  v8f acc[2][3];
#pragma unroll
  for (int h2 = 0; h2 < 2; ++h2)
#pragma unroll
    for (int t = 0; t < 3; ++t) acc[h2][t] = v8f{0, 0, 0, 0, 0, 0, 0, 0};

  for (int tap = 0; tap < ntap; ++tap) {
    int ky, kx; convt_tap(p, tap, ky, kx);
    const int di = (ky == 2) ? 1 : 0;
    const int dj = (kx == 2) ? 1 : 0;
    const u16* pa[2];
    bool ok[2];
#pragma unroll
    for (int h2 = 0; h2 < 2; ++h2) {
      const int iy = ii[h2] + di, ix = jj[h2] + dj;
      ok[h2] = (iy < 128) && (ix < 128);
      pa[h2] = X + ((size_t)((nn[h2] * 128 + iy) * 128 + ix)) * 512 + kg;
    }
    const int kbase = tap << 9;
    for (int cin0 = 0; cin0 < 512; cin0 += 32) {
      FragAB a[2];
#pragma unroll
      for (int h2 = 0; h2 < 2; ++h2) {
        if (ok[h2]) {
          const u16* pp = pa[h2] + cin0;
          a[h2].q[0] = *(const Q4*)pp;
          a[h2].q[1] = *(const Q4*)(pp + 16);
        } else { a[h2].q[0] = zero_q4(); a[h2].q[1] = zero_q4(); }
      }
#pragma unroll
      for (int nt = 0; nt < 3; ++nt) {
        FragAB b;
        const u16* pb = BT + offE + (size_t)(nt * 16 + bcol) * Kp + kbase + cin0 + bko;
        b.q[0] = *(const Q4*)pb;
        b.q[1] = *(const Q4*)(pb + 16);
        acc[0][nt] = wmma_bf16(a[0].v, b.v, acc[0][nt]);
        acc[1][nt] = wmma_bf16(a[1].v, b.v, acc[1][nt]);
      }
    }
  }
#pragma unroll
  for (int h2 = 0; h2 < 2; ++h2)
#pragma unroll
    for (int nt = 0; nt < 3; ++nt) {
      const int c = nt * 16 + bcol;
      if (c >= 40) continue;                    // padding columns
      const float bb = (c < 32) ? rgb[c] : hmb[c - 32];
#pragma unroll
      for (int v = 0; v < 8; ++v) {
        const int mm = mtile * 32 + h2 * 16 + v + half * 8;
        const int n2 = mm >> 14;
        const int rem2 = mm & 16383;
        const int oy2 = 2 * (rem2 >> 7) + py;
        const int ox2 = 2 * (rem2 & 127) + px;
        const float val = acc[h2][nt][v] + bb;
        if (c < 32)
          R[(((size_t)(n2 * 32 + c)) << 16) + oy2 * 256 + ox2] = val;
        else
          HMo[(((size_t)(n2 * 8 + (c - 32))) << 16) + oy2 * 256 + ox2] = val;
      }
    }
}

// ---------------------------------------------------------------------------
// BatchNorm (training-mode batch stats)
// ---------------------------------------------------------------------------
__global__ __launch_bounds__(256) void k_bn_stats(const float* __restrict__ x, int M, int C,
                                                  float* __restrict__ mean, float* __restrict__ rstd) {
  __shared__ float ss[256], sq[256];
  int c = blockIdx.x, t = threadIdx.x;
  float s = 0.f, q = 0.f;
  for (int m = t; m < M; m += 256) {
    float v = x[(size_t)m * C + c];
    s += v; q += v * v;
  }
  ss[t] = s; sq[t] = q;
  __syncthreads();
  for (int d = 128; d > 0; d >>= 1) {
    if (t < d) { ss[t] += ss[t + d]; sq[t] += sq[t + d]; }
    __syncthreads();
  }
  if (t == 0) {
    float mu = ss[0] / (float)M;
    float var = sq[0] / (float)M - mu * mu;
    mean[c] = mu; rstd[c] = rsqrtf(var + 1e-5f);
  }
}

__global__ __launch_bounds__(256) void k_bn_stats_nchw(const float* __restrict__ x, int Nb, int C,
                                                       int HW, float* __restrict__ mean,
                                                       float* __restrict__ rstd) {
  __shared__ float ss[256], sq[256];
  int c = blockIdx.x, t = threadIdx.x;
  float s = 0.f, q = 0.f;
  for (int nb = 0; nb < Nb; ++nb) {
    const float* pl = x + ((size_t)(nb * C + c)) * HW;
    for (int m = t; m < HW; m += 256) { float v = pl[m]; s += v; q += v * v; }
  }
  ss[t] = s; sq[t] = q;
  __syncthreads();
  for (int d = 128; d > 0; d >>= 1) {
    if (t < d) { ss[t] += ss[t + d]; sq[t] += sq[t + d]; }
    __syncthreads();
  }
  if (t == 0) {
    float M = (float)Nb * (float)HW;
    float mu = ss[0] / M;
    float var = sq[0] / M - mu * mu;
    mean[c] = mu; rstd[c] = rsqrtf(var + 1e-5f);
  }
}

__global__ __launch_bounds__(256) void k_bn_apply_bf16(const float* __restrict__ x,
                                                       u16* __restrict__ y,
                                                       const float* __restrict__ mean,
                                                       const float* __restrict__ rstd,
                                                       const float* __restrict__ g,
                                                       const float* __restrict__ be,
                                                       size_t total, int C, int do_hsw) {
  size_t i = (size_t)blockIdx.x * blockDim.x + threadIdx.x;
  if (i >= total) return;
  int c = (int)(i % C);
  float v = (x[i] - mean[c]) * rstd[c] * g[c] + be[c];
  if (do_hsw) v = hswish(v);
  y[i] = f2bf(v);
}

__global__ __launch_bounds__(256) void k_bn_hsw_nchw(float* __restrict__ x,
                                                     const float* __restrict__ mean,
                                                     const float* __restrict__ rstd,
                                                     const float* __restrict__ g,
                                                     const float* __restrict__ be,
                                                     int C, int HW, size_t total) {
  size_t i = (size_t)blockIdx.x * blockDim.x + threadIdx.x;
  if (i >= total) return;
  int c = (int)((i / HW) % C);
  float v = (x[i] - mean[c]) * rstd[c] * g[c] + be[c];
  x[i] = hswish(v);
}

// ---------------------------------------------------------------------------
// Classify tail: 3x3 conv 32->1 + flatten(1024) @ dw.T + db   (per batch)
// ---------------------------------------------------------------------------
__global__ __launch_bounds__(256) void k_classify(const u16* __restrict__ C1B,
                                                  const float* __restrict__ w2,
                                                  const float* __restrict__ b2,
                                                  const float* __restrict__ dw,
                                                  const float* __restrict__ db,
                                                  float* __restrict__ cls) {
  __shared__ float wsm[288];
  __shared__ float r0[256], r1[256];
  int b = blockIdx.x, t = threadIdx.x;
  for (int i = t; i < 288; i += 256) wsm[i] = w2[i];
  __syncthreads();
  float d0 = 0.f, d1 = 0.f;
  for (int pp = 0; pp < 4; ++pp) {
    int pix = t * 4 + pp;
    int y = pix >> 5, x = pix & 31;
    float acc = b2[0];
    for (int ky = 0; ky < 3; ++ky) {
      int iy = y + ky - 1;
      if (iy < 0 || iy >= 32) continue;
      for (int kx = 0; kx < 3; ++kx) {
        int ix = x + kx - 1;
        if (ix < 0 || ix >= 32) continue;
        const u16* pin = C1B + ((size_t)((b * 32 + iy) * 32 + ix)) * 32;
        for (int c = 0; c < 32; ++c)
          acc += bf2f(pin[c]) * wsm[c * 9 + ky * 3 + kx];
      }
    }
    d0 += acc * dw[pix];
    d1 += acc * dw[1024 + pix];
  }
  r0[t] = d0; r1[t] = d1;
  __syncthreads();
  for (int d = 128; d > 0; d >>= 1) {
    if (t < d) { r0[t] += r0[t + d]; r1[t] += r1[t + d]; }
    __syncthreads();
  }
  if (t == 0) { cls[b * 2 + 0] = r0[0] + db[0]; cls[b * 2 + 1] = r1[0] + db[1]; }
}

// ---------------------------------------------------------------------------
// Final regression conv: 3x3, 32 -> 2, pad 1 on [8,*,256,256] (R is BN+hsw'ed)
// ---------------------------------------------------------------------------
__global__ __launch_bounds__(256) void k_regconv(const float* __restrict__ R,
                                                 const float* __restrict__ w,
                                                 const float* __restrict__ b,
                                                 float* __restrict__ out) {
  __shared__ float ws[576];
  for (int i = threadIdx.x; i < 576; i += 256) ws[i] = w[i];
  __syncthreads();
  int idx = blockIdx.x * blockDim.x + threadIdx.x;   // 8*65536
  int nidx = idx >> 16;
  int s = idx & 65535;
  int y = s >> 8, x = s & 255;
  float a0 = b[0], a1 = b[1];
  for (int c = 0; c < 32; ++c) {
    const float* plane = R + (((size_t)(nidx * 32 + c)) << 16);
    for (int ky = 0; ky < 3; ++ky) {
      int yy = y + ky - 1;
      if (yy < 0 || yy >= 256) continue;
      for (int kx = 0; kx < 3; ++kx) {
        int xx = x + kx - 1;
        if (xx < 0 || xx >= 256) continue;
        float v = plane[yy * 256 + xx];
        a0 += v * ws[c * 9 + ky * 3 + kx];
        a1 += v * ws[288 + c * 9 + ky * 3 + kx];
      }
    }
  }
  out[(((size_t)(nidx * 2 + 0)) << 16) + s] = a0;
  out[(((size_t)(nidx * 2 + 1)) << 16) + s] = a1;
}

// ---------------------------------------------------------------------------
// Per-(batch,channel) spatial argmax (first-occurrence tie-break, as jnp.argmax)
// ---------------------------------------------------------------------------
__global__ __launch_bounds__(256) void k_argmax(const float* __restrict__ HMo,
                                                int* __restrict__ bidx) {
  __shared__ float bv[256];
  __shared__ int   bi[256];
  int bc = blockIdx.x, t = threadIdx.x;
  const float* pl = HMo + ((size_t)bc << 16);
  float mv = -3.402823466e38f;
  int mi = 0x7fffffff;
  for (int s = t; s < 65536; s += 256) {
    float v = pl[s];
    if (v > mv || (v == mv && s < mi)) { mv = v; mi = s; }
  }
  bv[t] = mv; bi[t] = mi;
  __syncthreads();
  for (int d = 128; d > 0; d >>= 1) {
    if (t < d) {
      if (bv[t + d] > bv[t] || (bv[t + d] == bv[t] && bi[t + d] < bi[t])) {
        bv[t] = bv[t + d]; bi[t] = bi[t + d];
      }
    }
    __syncthreads();
  }
  if (t == 0) bidx[bc] = bi[0];
}

__global__ void k_decode(const int* __restrict__ bidx, const float* __restrict__ REG,
                         const float* __restrict__ cls, float* __restrict__ out) {
  int t = threadIdx.x;
  if (t >= 16) return;
  int b = t >> 1, page = t & 1;
  float* row = out + (size_t)(b * 2 + page) * 9;
  bool valid = cls[b * 2 + page] > 0.6f;
  if (!valid) {
    for (int j2 = 0; j2 < 9; ++j2) row[j2] = -1.0f;
    return;
  }
  row[0] = (float)page;
  for (int corner = 0; corner < 4; ++corner) {
    int ch = page * 4 + corner;
    int idx = bidx[b * 8 + ch];
    int y = idx >> 8, x = idx & 255;
    float offx = REG[(((size_t)(b * 2 + 0)) << 16) + idx];
    float offy = REG[(((size_t)(b * 2 + 1)) << 16) + idx];
    row[1 + corner] = fminf(fmaxf((offx + (float)x) * (1.0f / 256.0f), 0.0f), 1.0f);
    row[5 + corner] = fminf(fmaxf((offy + (float)y) * (1.0f / 256.0f), 0.0f), 1.0f);
  }
}

// ---------------------------------------------------------------------------
extern "C" void kernel_launch(void* const* d_in, const int* in_sizes, int n_in,
                              void* d_out, int out_size, void* d_ws, size_t ws_size,
                              hipStream_t stream) {
  (void)in_sizes; (void)n_in; (void)out_size; (void)ws_size;
  const float* feature = (const float*)d_in[0];
  const float* ch_w1 = (const float*)d_in[1];  const float* ch_b1 = (const float*)d_in[2];
  const float* ch_w2 = (const float*)d_in[3];  const float* ch_b2 = (const float*)d_in[4];
  const float* ch_w3 = (const float*)d_in[5];  const float* ch_b3 = (const float*)d_in[6];
  const float* ch_g1 = (const float*)d_in[7];  const float* ch_be1 = (const float*)d_in[8];
  const float* ch_g2 = (const float*)d_in[9];  const float* ch_be2 = (const float*)d_in[10];
  const float* cl_w1 = (const float*)d_in[11]; const float* cl_b1 = (const float*)d_in[12];
  const float* cl_w2 = (const float*)d_in[13]; const float* cl_b2 = (const float*)d_in[14];
  const float* cl_g  = (const float*)d_in[15]; const float* cl_be = (const float*)d_in[16];
  const float* cl_dw = (const float*)d_in[17]; const float* cl_db = (const float*)d_in[18];
  const float* rg_tw = (const float*)d_in[19]; const float* rg_tb = (const float*)d_in[20];
  const float* rg_g  = (const float*)d_in[21]; const float* rg_be = (const float*)d_in[22];
  const float* rg_w  = (const float*)d_in[23]; const float* rg_b  = (const float*)d_in[24];
  const float* hm_tw = (const float*)d_in[25]; const float* hm_tb = (const float*)d_in[26];

  char* base = (char*)d_ws;
  size_t off = 0;
  auto take = [&](size_t bytes) -> char* {
    char* p = base + off;
    off = (off + bytes + 255) & ~(size_t)255;
    return p;
  };
  u16*   Fb  = (u16*)take((size_t)8 * 128 * 128 * 512 * 2);   // 134 MB (fits L2)
  u16*   W1  = (u16*)take((size_t)192 * 4608 * 2);
  u16*   W2  = (u16*)take((size_t)192 * 1728 * 2);
  u16*   W3  = (u16*)take((size_t)32 * 192 * 2);
  u16*   CW1 = (u16*)take((size_t)32 * 288 * 2);
  u16*   BT  = (u16*)take((size_t)48 * 4608 * 2);
  float* O1  = (float*)take((size_t)32768 * 192 * 4);
  u16*   O1B = (u16*)take((size_t)32768 * 192 * 2);
  float* O2  = (float*)take((size_t)8192 * 192 * 4);
  u16*   O2B = (u16*)take((size_t)8192 * 192 * 2);
  float* O3  = (float*)take((size_t)8192 * 32 * 4);
  u16*   HB  = (u16*)take((size_t)8192 * 32 * 2);
  float* C1  = (float*)take((size_t)8192 * 32 * 4);
  u16*   C1B = (u16*)take((size_t)8192 * 32 * 2);
  float* CLS = (float*)take(16 * 4);
  float* R   = (float*)take((size_t)8 * 32 * 65536 * 4);
  float* HMo = (float*)take((size_t)8 * 8 * 65536 * 4);
  float* REG = (float*)take((size_t)8 * 2 * 65536 * 4);
  float* ST  = (float*)take(1024 * 4);
  int*   IDX = (int*)take(64 * 4);

  float* s_m1 = ST + 0,   * s_r1 = ST + 192;
  float* s_m2 = ST + 384, * s_r2 = ST + 576;
  float* s_m3 = ST + 768, * s_r3 = ST + 800;
  float* s_mc = ST + 832, * s_rc = ST + 864;
  float* s_mr = ST + 896, * s_rr = ST + 928;

  dim3 B(256);

  // ---- precompute: bf16 feature + packed weights ----
  k_cvt_feature<<<65536, B, 0, stream>>>(feature, Fb);
  k_pack_w3x3<<<(884736 + 255) / 256, B, 0, stream>>>(ch_w1, W1, 192, 512);
  k_pack_w3x3<<<(331776 + 255) / 256, B, 0, stream>>>(ch_w2, W2, 192, 192);
  k_cvt_bf<<<(6144 + 255) / 256, B, 0, stream>>>(ch_w3, W3, 6144);
  k_pack_w3x3<<<(9216 + 255) / 256, B, 0, stream>>>(cl_w1, CW1, 32, 32);
  k_pack_bt<<<(221184 + 255) / 256, B, 0, stream>>>(rg_tw, hm_tw, BT);

  // ---- CommonHead ----
  // conv1: M=32768 -> 1024 Mtiles(32) x 3 ngroups = 3072 waves = 384 blocks
  k_conv_wmma<<<384, B, 0, stream>>>(Fb, W1, ch_b1, O1, 8, 128, 128, 512, 64, 64, 192, 3, 2, 1);
  k_bn_stats<<<192, B, 0, stream>>>(O1, 32768, 192, s_m1, s_r1);
  k_bn_apply_bf16<<<24576, B, 0, stream>>>(O1, O1B, s_m1, s_r1, ch_g1, ch_be1,
                                           (size_t)32768 * 192, 192, 0);
  // conv2: M=8192 -> 256 x 3 = 768 waves = 96 blocks
  k_conv_wmma<<<96, B, 0, stream>>>(O1B, W2, ch_b2, O2, 8, 64, 64, 192, 32, 32, 192, 3, 2, 1);
  k_bn_stats<<<192, B, 0, stream>>>(O2, 8192, 192, s_m2, s_r2);
  k_bn_apply_bf16<<<6144, B, 0, stream>>>(O2, O2B, s_m2, s_r2, ch_g1, ch_be1,
                                          (size_t)8192 * 192, 192, 0);   // bn1 reused!
  // conv3 (1x1): M=8192 -> 256 waves = 32 blocks
  k_conv_wmma<<<32, B, 0, stream>>>(O2B, W3, ch_b3, O3, 8, 32, 32, 192, 32, 32, 32, 1, 1, 0);
  k_bn_stats<<<32, B, 0, stream>>>(O3, 8192, 32, s_m3, s_r3);
  k_bn_apply_bf16<<<1024, B, 0, stream>>>(O3, HB, s_m3, s_r3, ch_g2, ch_be2,
                                          (size_t)8192 * 32, 32, 1);

  // ---- PageClassifyNet ----
  k_conv_wmma<<<32, B, 0, stream>>>(HB, CW1, cl_b1, C1, 8, 32, 32, 32, 32, 32, 32, 3, 1, 1);
  k_bn_stats<<<32, B, 0, stream>>>(C1, 8192, 32, s_mc, s_rc);
  k_bn_apply_bf16<<<1024, B, 0, stream>>>(C1, C1B, s_mc, s_rc, cl_g, cl_be,
                                          (size_t)8192 * 32, 32, 1);
  k_classify<<<8, B, 0, stream>>>(C1B, cl_w2, cl_b2, cl_dw, cl_db, CLS);

  // ---- ConvTranspose branches (fused rg + hm) ----
  // 4 parities x 4096 Mtiles(32) = 16384 waves = 2048 blocks
  k_convt_wmma<<<2048, B, 0, stream>>>(Fb, BT, rg_tb, hm_tb, R, HMo);
  k_bn_stats_nchw<<<32, B, 0, stream>>>(R, 8, 32, 65536, s_mr, s_rr);
  k_bn_hsw_nchw<<<65536, B, 0, stream>>>(R, s_mr, s_rr, rg_g, rg_be, 32, 65536,
                                         (size_t)8 * 32 * 65536);
  k_regconv<<<2048, B, 0, stream>>>(R, rg_w, rg_b, REG);

  // ---- decode ----
  k_argmax<<<64, B, 0, stream>>>(HMo, IDX);
  k_decode<<<1, 64, 0, stream>>>(IDX, REG, CLS, (float*)d_out);
}
